// BayesianSkipgram_51419348468081
// MI455X (gfx1250) — matrix operand order repaired and checked by reading, not compiled, not run on hardware
//
#include <hip/hip_runtime.h>

// MI455X / gfx1250 — fully fused Bayesian skip-gram KL kernel.
// One block = 16 batch rows, 8 wave32s (one 16-wide slice of D per wave).
// Stage 1: gathered GEMM [16*(C+1), 256] x [256, 128] via v_wmma_f32_16x16x32_bf16,
//          ping-pong LDS A-tiles (1 barrier/row-kind), relu+sum -> h[16,256] (bf16, LDS).
// Stage 2: mu/sigma GEMMs from the LDS h-tile + closed-form diagonal-Gaussian KL.

typedef __attribute__((ext_vector_type(16))) __bf16 v16bf;
typedef __attribute__((ext_vector_type(2)))  __bf16 v2bf;
typedef __attribute__((ext_vector_type(8)))  float  v8f;

constexpr int E = 256;   // embedding size (K dim of stage-1 GEMM, and 2D)
constexpr int D = 128;   // latent dim
constexpr int CWIN = 10; // context window

// f32 pair -> packed bf16 dword (lowers to v_cvt_pk_bf16_f32).
__device__ __forceinline__ unsigned int packbf(float lo, float hi) {
  v2bf t;
  t.x = (__bf16)lo;
  t.y = (__bf16)hi;
  return __builtin_bit_cast(unsigned int, t);
}

union BFrag {
  v16bf v;
  uint4 q[2];
};

// 16 consecutive f32 at wrow+koff -> bf16 B-fragment
// (B-matrix 32x16 bf16: lane holds column n = lane%16, element j <-> K = 16*(lane/16) + j).
__device__ __forceinline__ v16bf load_wrow_frag(const float* __restrict__ wrow, int koff) {
  const float4* p = reinterpret_cast<const float4*>(wrow + koff);
  float4 a = p[0], b = p[1], c = p[2], d = p[3];
  BFrag f;
  f.q[0] = make_uint4(packbf(a.x, a.y), packbf(a.z, a.w), packbf(b.x, b.y), packbf(b.z, b.w));
  f.q[1] = make_uint4(packbf(c.x, c.y), packbf(c.z, c.w), packbf(d.x, d.y), packbf(d.z, d.w));
  return f.v;
}

// A-matrix 16x32 bf16 fragment from an LDS tile (rows = 128 dwords of packed bf16).
// Lane holds row m = lane%16; j=0..15 -> k = 32kc + 16*khalf + j.
__device__ __forceinline__ v16bf load_a_frag(const unsigned int* ldsA, int row, int kc, int khalf) {
  const uint4* p = reinterpret_cast<const uint4*>(ldsA + row * 128 + kc * 16 + khalf * 8);
  BFrag f;
  f.q[0] = p[0];
  f.q[1] = p[1];
  return f.v;
}

__global__ void __launch_bounds__(256)
bsg_fused_kernel(const int* __restrict__ x, const int* __restrict__ ctx,
                 const float* __restrict__ W_emb, const float* __restrict__ M_w,
                 const float* __restrict__ M_b, const float* __restrict__ U_w,
                 const float* __restrict__ U_b, const float* __restrict__ W_w,
                 const float* __restrict__ W_b, const float* __restrict__ pmu,
                 const float* __restrict__ psig, float* __restrict__ out) {
  // Ping-pong gathered-embedding tiles; buffer 0 is reused for the bf16 h-tile.
  __shared__ __attribute__((aligned(16))) unsigned int ldsA[2][16 * 128]; // 2 x 8 KB
  __shared__ float klpart[8][16];
  __shared__ int   xi[16];

  const int tid   = threadIdx.x;
  const int b0    = blockIdx.x * 16;
  const int wave  = tid >> 5;
  const int lane  = tid & 31;
  const int nlo   = lane & 15;
  const int khalf = lane >> 4;
  const int drow  = wave * 16 + nlo;          // output column (d) this lane owns

  if (tid < 16) xi[tid] = x[b0 + tid];

  // ---- Stage-1 B fragments: this wave's 16 M_w rows, all 8 K-chunks, in VGPRs ----
  v16bf bfrag[8];
  {
    const float* wrow = M_w + (size_t)drow * E;
#pragma unroll
    for (int kc = 0; kc < 8; ++kc)
      bfrag[kc] = load_wrow_frag(wrow, kc * 32 + khalf * 16);
  }
  const float bias = M_b[drow];

  float h1[8], h2[8];
#pragma unroll
  for (int p = 0; p < 8; ++p) { h1[p] = 0.f; h2[p] = 0.f; }

  const int arow = tid >> 4;          // this thread stages row arow,
  const int akb  = (tid & 15) * 16;   // 16 floats starting at column akb

  // Prologue: stage the r=0 (target word) tile into buffer 0.
  {
    int idx = x[b0 + arow];
    const float4* src = reinterpret_cast<const float4*>(W_emb + (size_t)idx * E + akb);
    float4 f0 = src[0], f1 = src[1], f2 = src[2], f3 = src[3];
    uint4* dst = reinterpret_cast<uint4*>(&ldsA[0][0] + arow * 128 + (akb >> 1));
    dst[0] = make_uint4(packbf(f0.x, f0.y), packbf(f0.z, f0.w), packbf(f1.x, f1.y), packbf(f1.z, f1.w));
    dst[1] = make_uint4(packbf(f2.x, f2.y), packbf(f2.z, f2.w), packbf(f3.x, f3.y), packbf(f3.z, f3.w));
  }
  __syncthreads();

  // Prior gathers issued early: latency hides under the whole GEMM phase.
  float m0v[8], s0v[8];
#pragma unroll
  for (int p = 0; p < 8; ++p) {
    int xv = xi[p + khalf * 8];
    m0v[p] = pmu [(size_t)xv * D + drow];
    s0v[p] = psig[(size_t)xv * D + drow];
  }

  // ---- Stage-1 main loop: one barrier per row-kind via ping-pong tiles ----
  for (int r = 0; r <= CWIN; ++r) {
    const unsigned int* rbuf = &ldsA[r & 1][0];

    // Prefetch next row-kind's gather while this tile's WMMAs run.
    float4 f0, f1, f2, f3;
    if (r < CWIN) {
      int idx = ctx[(size_t)(b0 + arow) * CWIN + r];
      const float4* src = reinterpret_cast<const float4*>(W_emb + (size_t)idx * E + akb);
      f0 = src[0]; f1 = src[1]; f2 = src[2]; f3 = src[3];
    }

    // Two independent accumulation chains over the 8 K-chunks.
    v8f acc0 = {}, acc1 = {};
#pragma unroll
    for (int kc = 0; kc < 8; kc += 2) {
      v16bf a0 = load_a_frag(rbuf, nlo, kc,     khalf);
      v16bf a1 = load_a_frag(rbuf, nlo, kc + 1, khalf);
      acc0 = __builtin_amdgcn_wmma_f32_16x16x32_bf16(false, a0, false, bfrag[kc],     (short)0, acc0, false, false);
      acc1 = __builtin_amdgcn_wmma_f32_16x16x32_bf16(false, a1, false, bfrag[kc + 1], (short)0, acc1, false, false);
    }

    // Stage next tile into the other buffer (its prior readers finished last iter).
    if (r < CWIN) {
      uint4* dst = reinterpret_cast<uint4*>(&ldsA[(r + 1) & 1][0] + arow * 128 + (akb >> 1));
      dst[0] = make_uint4(packbf(f0.x, f0.y), packbf(f0.z, f0.w), packbf(f1.x, f1.y), packbf(f1.z, f1.w));
      dst[1] = make_uint4(packbf(f2.x, f2.y), packbf(f2.z, f2.w), packbf(f3.x, f3.y), packbf(f3.z, f3.w));
    }
    __syncthreads();

    // relu + (C * target | sum over context)
#pragma unroll
    for (int p = 0; p < 8; ++p) {
      float v = acc0[p] + acc1[p] + bias;
      v = v > 0.f ? v : 0.f;
      if (r == 0) h1[p] = (float)CWIN * v;
      else        h2[p] += v;
    }
  }

  // ---- Write the h[16, 256] tile into LDS buffer 0 as bf16 (stage-2 A matrix) ----
  // All waves passed the final loop barrier after their last reads of buffer 0.
  {
    __bf16* ldsH = reinterpret_cast<__bf16*>(&ldsA[0][0]);
#pragma unroll
    for (int p = 0; p < 8; ++p) {
      int row = p + khalf * 8;
      ldsH[row * 256 + drow]       = (__bf16)h1[p];
      ldsH[row * 256 + 128 + drow] = (__bf16)h2[p];
    }
  }

  // Stage-2 projection fragments (U_w and W_w rows for this wave's d-slice).
  v16bf ufrag[8], wfrag[8];
  {
    const float* urow = U_w + (size_t)drow * 256;
    const float* wrow = W_w + (size_t)drow * 256;
#pragma unroll
    for (int kc = 0; kc < 8; ++kc) {
      ufrag[kc] = load_wrow_frag(urow, kc * 32 + khalf * 16);
      wfrag[kc] = load_wrow_frag(wrow, kc * 32 + khalf * 16);
    }
  }
  __syncthreads();

  // ---- Stage 2: mu / sigma-pre GEMMs (two independent WMMA chains share A) ----
  v8f mu = {}, sg = {};
#pragma unroll
  for (int kc = 0; kc < 8; ++kc) {
    v16bf afrag = load_a_frag(&ldsA[0][0], nlo, kc, khalf);
    mu = __builtin_amdgcn_wmma_f32_16x16x32_bf16(false, afrag, false, ufrag[kc], (short)0, mu, false, false);
    sg = __builtin_amdgcn_wmma_f32_16x16x32_bf16(false, afrag, false, wfrag[kc], (short)0, sg, false, false);
  }
  const float ub = U_b[drow];
  const float wb = W_b[drow];

  // ---- Per-element KL terms ----
  float term[8];
#pragma unroll
  for (int p = 0; p < 8; ++p) {
    float m  = mu[p] + ub;
    float sp = sg[p] + wb;
    float sig = fmaxf(sp, 0.f) + __logf(1.f + __expf(-fabsf(sp)));  // softplus
    float dm  = m - m0v[p];
    term[p] = (s0v[p] + dm * dm) / sig + __logf(sig) - __logf(s0v[p]);
  }

  // Rows 0-7 live in lanes 0-15, rows 8-15 in lanes 16-31:
  // xor-shuffle reduce over each 16-lane half, then deterministic cross-wave sum.
#pragma unroll
  for (int p = 0; p < 8; ++p) {
#pragma unroll
    for (int m = 1; m < 16; m <<= 1)
      term[p] += __shfl_xor(term[p], m, 32);
  }
  if (nlo == 0) {
#pragma unroll
    for (int p = 0; p < 8; ++p)
      klpart[wave][p + khalf * 8] = term[p];
  }
  __syncthreads();

  if (tid < 16) {
    float s = 0.f;
#pragma unroll
    for (int w = 0; w < 8; ++w) s += klpart[w][tid];   // fixed order -> deterministic
    out[b0 + tid] = 0.5f * (s - (float)D);
  }
}

extern "C" void kernel_launch(void* const* d_in, const int* in_sizes, int n_in,
                              void* d_out, int out_size, void* d_ws, size_t ws_size,
                              hipStream_t stream) {
  const int*   x     = (const int*)d_in[0];
  const int*   ctx   = (const int*)d_in[1];
  const float* W_emb = (const float*)d_in[2];
  const float* M_w   = (const float*)d_in[3];
  const float* M_b   = (const float*)d_in[4];
  const float* U_w   = (const float*)d_in[5];
  const float* U_b   = (const float*)d_in[6];
  const float* W_w   = (const float*)d_in[7];
  const float* W_b   = (const float*)d_in[8];
  const float* pmu   = (const float*)d_in[9];
  const float* psig  = (const float*)d_in[10];
  float* out = (float*)d_out;
  (void)d_ws; (void)ws_size; (void)n_in; (void)out_size;

  const int B = in_sizes[0];          // 8192
  dim3 block(256);                    // 8 wave32s per workgroup
  dim3 grid(B / 16);                  // one 16-row batch tile per workgroup

  bsg_fused_kernel<<<grid, block, 0, stream>>>(x, ctx, W_emb, M_w, M_b,
                                               U_w, U_b, W_w, W_b, pmu, psig, out);
}
